// CausalAttention_69810398429498
// MI455X (gfx1250) — compile-verified, matrix operand order checked by reading, not run on hardware
//
#include <hip/hip_runtime.h>
#include <hip/hip_bf16.h>

// ---------------------------------------------------------------------------
// CausalAttention for MI455X (gfx1250): bf16 WMMA pipeline, fp32 accumulate.
//   B=4, T=2048, D_IN=D_OUT=1024, single head, mask = tril(diag=1),
//   scale = 1/sqrt(1024) = 1/32.
// Register double-buffered GEMMs: wave tile 32x64, block tile 128x128.
// ---------------------------------------------------------------------------

typedef __bf16 bf16_t;
typedef __attribute__((ext_vector_type(16))) __bf16 v16bf;
typedef __attribute__((ext_vector_type(8)))  float  v8f;
typedef __attribute__((ext_vector_type(4)))  int    i4;
typedef __attribute__((ext_vector_type(4)))  float  v4f;
typedef __attribute__((ext_vector_type(4)))  unsigned short v4u16;

#define T_SEQ  2048
#define D_DIM  1024
#define NBATCH 4

union FragBF { v16bf v; i4 q[2]; };

// float -> bf16 round-to-nearest-even (manual, backend-safe)
__device__ __forceinline__ bf16_t f2bf(float x) {
  unsigned u = __builtin_bit_cast(unsigned, x);
  unsigned r = u + 0x7fffu + ((u >> 16) & 1u);
  unsigned short h = (unsigned short)(r >> 16);
  return __builtin_bit_cast(bf16_t, h);
}
__device__ __forceinline__ unsigned short f2bf_u16(float x) {
  unsigned u = __builtin_bit_cast(unsigned, x);
  unsigned r = u + 0x7fffu + ((u >> 16) & 1u);
  return (unsigned short)(r >> 16);
}

// Load a 16x32 bf16 WMMA A-fragment (also used for B via B^T stored row-major,
// ld in elements, 16B-aligned rows). ISA 7.12.2 16-bit A layout:
//   lane&15 = matrix row; lane>=16 -> K offset +8; elems 8..15 -> K offset +16.
__device__ __forceinline__ v16bf load_frag(const bf16_t* base, int ld) {
  int lane = threadIdx.x & 31;
  int row  = lane & 15;
  int koff = (lane >> 4) << 3;            // 0 or 8
  const bf16_t* p = base + (size_t)row * ld + koff;
  FragBF f;
  f.q[0] = *(const i4*)(p);               // K = koff .. koff+7
  f.q[1] = *(const i4*)(p + 16);          // K = koff+16 .. koff+23
  return f.v;
}

__device__ __forceinline__ v8f wmma_bf16(v16bf a, v16bf b, v8f c) {
  return __builtin_amdgcn_wmma_f32_16x16x32_bf16(
      /*neg_a=*/false, a, /*neg_b=*/false, b,
      /*c_mod=*/(short)0, c, /*reuse_a=*/false, /*reuse_b=*/false);
}

// ---- 32x64 per-wave GEMM core, register double-buffered over K ------------
__device__ __forceinline__ void load_ab(const bf16_t* A, const bf16_t* B,
                                        int lda, int ldb, int kk,
                                        v16bf a[2], v16bf b[4]) {
  a[0] = load_frag(A + kk, lda);
  a[1] = load_frag(A + (size_t)16 * lda + kk, lda);
#pragma unroll
  for (int t = 0; t < 4; ++t)
    b[t] = load_frag(B + (size_t)(t * 16) * ldb + kk, ldb);
}

__device__ __forceinline__ void mma_step(const v16bf a[2], const v16bf b[4],
                                         v8f acc[2][4]) {
#pragma unroll
  for (int m = 0; m < 2; ++m)
#pragma unroll
    for (int t = 0; t < 4; ++t)
      acc[m][t] = wmma_bf16(a[m], b[t], acc[m][t]);
}

// A: rows m0..m0+31 (lda), B^T: rows n0..n0+63 (ldb), K = kend (mult of 32)
__device__ __forceinline__ void gemm_pipe(const bf16_t* A, const bf16_t* B,
                                          int lda, int ldb, int kend,
                                          v8f acc[2][4]) {
  v16bf aA[2], bA[4], aB[2], bB[4];
  int nsteps = kend >> 5;                  // >= 1
  load_ab(A, B, lda, ldb, 0, aA, bA);
  int s = 1;
  for (; s + 1 < nsteps; s += 2) {
    load_ab(A, B, lda, ldb, s << 5, aB, bB);      // prefetch step s
    mma_step(aA, bA, acc);                        // consume step s-1
    load_ab(A, B, lda, ldb, (s + 1) << 5, aA, bA);
    mma_step(aB, bB, acc);
  }
  if (s < nsteps) {                               // one trailing step pair
    load_ab(A, B, lda, ldb, s << 5, aB, bB);
    mma_step(aA, bA, acc);
    mma_step(aB, bB, acc);
  } else {
    mma_step(aA, bA, acc);
  }
}

// ---------------------------------------------------------------------------
// Kernel 1: fp32 -> bf16 conversion (x4 vectorized)
// ---------------------------------------------------------------------------
__global__ void cvt_kernel(const float* __restrict__ in, bf16_t* __restrict__ out,
                           int n4) {
  int i = blockIdx.x * blockDim.x + threadIdx.x;
  if (i >= n4) return;
  v4f v = *(const v4f*)(in + (size_t)i * 4);
  v4u16 o;
  o.x = f2bf_u16(v.x); o.y = f2bf_u16(v.y);
  o.z = f2bf_u16(v.z); o.w = f2bf_u16(v.w);
  *(v4u16*)(out + (size_t)i * 4) = o;
}

// ---------------------------------------------------------------------------
// Kernel 2: projection GEMM  Y[M=8192, N=1024] = Xb[M,K] * W[N,K]^T
// 8 waves: wave>>1 -> M strip (32 rows), wave&1 -> N strip (64 cols).
// mode 0: Y row-major bf16; mode 2: per-batch transposed store [4][D][T] (V).
// ---------------------------------------------------------------------------
__global__ __launch_bounds__(256)
void qkv_proj_kernel(const bf16_t* __restrict__ X, const bf16_t* __restrict__ W,
                     bf16_t* __restrict__ Y, int mode) {
  const int K = D_DIM;
  int wave = threadIdx.x >> 5;
  int m0 = blockIdx.x * 128 + (wave >> 1) * 32;
  int n0 = blockIdx.y * 128 + (wave & 1) * 64;
  v8f acc[2][4] = {};
  gemm_pipe(X + (size_t)m0 * K, W + (size_t)n0 * K, K, K, K, acc);

  int lane = threadIdx.x & 31;
  int n  = lane & 15;
  int mb = (lane >> 4) << 3;
#pragma unroll
  for (int m = 0; m < 2; ++m)
#pragma unroll
    for (int t = 0; t < 4; ++t) {
      int col = n0 + t * 16 + n;
#pragma unroll
      for (int i = 0; i < 8; ++i) {
        int row = m0 + m * 16 + mb + i;
        bf16_t v = f2bf(acc[m][t][i]);
        if (mode == 2) {
          int bb = row >> 11;               // row / T_SEQ
          int tr = row & (T_SEQ - 1);
          Y[(size_t)bb * D_DIM * T_SEQ + (size_t)col * T_SEQ + tr] = v;
        } else {
          Y[(size_t)row * D_DIM + col] = v;
        }
      }
    }
}

// ---------------------------------------------------------------------------
// Kernel 3: scores  S[b,q,k] = (Q[b,q,:] . K[b,k,:]) / 32, causal mask k<=q+1
// ---------------------------------------------------------------------------
__global__ __launch_bounds__(256)
void scores_kernel(const bf16_t* __restrict__ Q, const bf16_t* __restrict__ Kt,
                   float* __restrict__ S) {
  int bz = blockIdx.z;
  int wave = threadIdx.x >> 5;
  int q0 = blockIdx.x * 128 + (wave >> 1) * 32;
  int k0 = blockIdx.y * 128 + (wave & 1) * 64;
  const bf16_t* Qb = Q  + (size_t)bz * T_SEQ * D_DIM;
  const bf16_t* Kb = Kt + (size_t)bz * T_SEQ * D_DIM;
  float* Sb = S + (size_t)bz * T_SEQ * T_SEQ;

  int lane = threadIdx.x & 31;
  int n  = lane & 15;
  int mb = (lane >> 4) << 3;

  // Block-uniform: fully above the causal band -> -inf fill, skip the GEMM.
  if (blockIdx.y * 128 > blockIdx.x * 128 + 128) {
#pragma unroll
    for (int m = 0; m < 2; ++m)
#pragma unroll
      for (int t = 0; t < 4; ++t)
#pragma unroll
        for (int i = 0; i < 8; ++i)
          Sb[(size_t)(q0 + m * 16 + mb + i) * T_SEQ + (k0 + t * 16 + n)] =
              -__builtin_inff();
    return;
  }

  v8f acc[2][4] = {};
  gemm_pipe(Qb + (size_t)q0 * D_DIM, Kb + (size_t)k0 * D_DIM,
            D_DIM, D_DIM, D_DIM, acc);

  const float scale = 0.03125f;  // 1/sqrt(1024)
#pragma unroll
  for (int m = 0; m < 2; ++m)
#pragma unroll
    for (int t = 0; t < 4; ++t) {
      int k = k0 + t * 16 + n;
#pragma unroll
      for (int i = 0; i < 8; ++i) {
        int q = q0 + m * 16 + mb + i;
        float v = acc[m][t][i] * scale;
        if (k > q + 1) v = -__builtin_inff();
        Sb[(size_t)q * T_SEQ + k] = v;
      }
    }
}

// ---------------------------------------------------------------------------
// Kernel 4: row softmax over S (fp32) -> P (bf16). One block per (b,q) row.
// Row (8 KB) staged in LDS via CDNA5 GLOBAL_LOAD_ASYNC_TO_LDS_B128 (ASYNCcnt),
// so global memory is read exactly once per row.
// ---------------------------------------------------------------------------
__global__ __launch_bounds__(256)
void softmax_kernel(const float* __restrict__ S, bf16_t* __restrict__ P) {
  __shared__ float srow[T_SEQ];
  __shared__ float red[8];
  size_t row = blockIdx.x;
  const float* s = S + row * T_SEQ;
  bf16_t* p = P + row * T_SEQ;
  int tid = threadIdx.x;

  // Stage 32 bytes per thread: LDS[srow + tid*32 .. +32) = s[tid*8 .. +8)
  {
    unsigned lds0 = (unsigned)(size_t)(void*)&srow[tid * 8];
    unsigned long long g0 = (unsigned long long)(size_t)(const void*)&s[tid * 8];
    asm volatile(
        "global_load_async_to_lds_b128 %0, %1, off\n\t"
        "global_load_async_to_lds_b128 %0, %1, off offset:16"
        :: "v"(lds0), "v"(g0) : "memory");
    asm volatile("s_wait_asynccnt 0x0" ::: "memory");
  }
  __syncthreads();

  float lmax = -__builtin_inff();
  for (int i = tid; i < T_SEQ; i += 256) lmax = fmaxf(lmax, srow[i]);
  for (int off = 16; off > 0; off >>= 1) lmax = fmaxf(lmax, __shfl_xor(lmax, off, 32));
  if ((tid & 31) == 0) red[tid >> 5] = lmax;
  __syncthreads();
  float m = fmaxf(fmaxf(fmaxf(red[0], red[1]), fmaxf(red[2], red[3])),
                  fmaxf(fmaxf(red[4], red[5]), fmaxf(red[6], red[7])));
  __syncthreads();

  float lsum = 0.f;
  for (int i = tid; i < T_SEQ; i += 256) lsum += __expf(srow[i] - m);
  for (int off = 16; off > 0; off >>= 1) lsum += __shfl_xor(lsum, off, 32);
  if ((tid & 31) == 0) red[tid >> 5] = lsum;
  __syncthreads();
  float inv = 1.0f / (red[0] + red[1] + red[2] + red[3] +
                      red[4] + red[5] + red[6] + red[7]);

  for (int i = tid; i < T_SEQ; i += 256) p[i] = f2bf(__expf(srow[i] - m) * inv);
}

// ---------------------------------------------------------------------------
// Kernel 5: O[b,q,d] = sum_k P[b,q,k] * V[b,k,d], using Vt[b][d][k] (= B^T).
// K-loop truncated at the causal limit per wave (P is exactly 0 beyond).
// ---------------------------------------------------------------------------
__global__ __launch_bounds__(256)
void pv_kernel(const bf16_t* __restrict__ P, const bf16_t* __restrict__ Vt,
               float* __restrict__ O) {
  int bz = blockIdx.z;
  int wave = threadIdx.x >> 5;
  int q0 = blockIdx.x * 128 + (wave >> 1) * 32;
  int d0 = blockIdx.y * 128 + (wave & 1) * 64;
  const bf16_t* Pb = P  + (size_t)bz * T_SEQ * T_SEQ;
  const bf16_t* Vb = Vt + (size_t)bz * D_DIM * T_SEQ;
  float* Ob = O + (size_t)bz * T_SEQ * D_DIM;

  // rows q0..q0+31 see keys <= q0+32 -> round up to 64 (q0 is 32-aligned)
  int kend = q0 + 64;
  if (kend > T_SEQ) kend = T_SEQ;

  v8f acc[2][4] = {};
  gemm_pipe(Pb + (size_t)q0 * T_SEQ, Vb + (size_t)d0 * T_SEQ,
            T_SEQ, T_SEQ, kend, acc);

  int lane = threadIdx.x & 31;
  int n  = lane & 15;
  int mb = (lane >> 4) << 3;
#pragma unroll
  for (int m = 0; m < 2; ++m)
#pragma unroll
    for (int t = 0; t < 4; ++t) {
      int d = d0 + t * 16 + n;
#pragma unroll
      for (int i = 0; i < 8; ++i) {
        int q = q0 + m * 16 + mb + i;
        Ob[(size_t)q * D_DIM + d] = acc[m][t][i];
      }
    }
}

// ---------------------------------------------------------------------------
// Host launcher. Workspace layout (MiB offsets), total 166 MiB:
//   Qb 0..16 | Kb 16..32 | Vt 32..48 | Xb 48..64 | Wq/Wk/Wv 64..70 |
//   S 70..134 (fp32) | P 134..166 (bf16)
// ---------------------------------------------------------------------------
extern "C" void kernel_launch(void* const* d_in, const int* in_sizes, int n_in,
                              void* d_out, int out_size, void* d_ws, size_t ws_size,
                              hipStream_t stream) {
  const float* x  = (const float*)d_in[0];
  const float* Wq = (const float*)d_in[1];
  const float* Wk = (const float*)d_in[2];
  const float* Wv = (const float*)d_in[3];
  float* out = (float*)d_out;

  char* ws = (char*)d_ws;
  const size_t MB = 1024u * 1024u;
  bf16_t* Qb  = (bf16_t*)(ws + 0 * MB);
  bf16_t* Kb  = (bf16_t*)(ws + 16 * MB);
  bf16_t* Vtb = (bf16_t*)(ws + 32 * MB);
  bf16_t* Xb  = (bf16_t*)(ws + 48 * MB);
  bf16_t* Wqb = (bf16_t*)(ws + 64 * MB);
  bf16_t* Wkb = (bf16_t*)(ws + 66 * MB);
  bf16_t* Wvb = (bf16_t*)(ws + 68 * MB);
  float*  S   = (float*)(ws + 70 * MB);
  bf16_t* P   = (bf16_t*)(ws + 134 * MB);

  const int NX = NBATCH * T_SEQ * D_DIM;  // 8388608
  const int NW = D_DIM * D_DIM;           // 1048576

  cvt_kernel<<<(NX / 4 + 255) / 256, 256, 0, stream>>>(x,  Xb,  NX / 4);
  cvt_kernel<<<(NW / 4 + 255) / 256, 256, 0, stream>>>(Wq, Wqb, NW / 4);
  cvt_kernel<<<(NW / 4 + 255) / 256, 256, 0, stream>>>(Wk, Wkb, NW / 4);
  cvt_kernel<<<(NW / 4 + 255) / 256, 256, 0, stream>>>(Wv, Wvb, NW / 4);

  dim3 blk(256);
  dim3 gp(64, 8);  // M=8192/128, N=1024/128
  qkv_proj_kernel<<<gp, blk, 0, stream>>>(Xb, Wqb, Qb,  0);
  qkv_proj_kernel<<<gp, blk, 0, stream>>>(Xb, Wkb, Kb,  0);
  qkv_proj_kernel<<<gp, blk, 0, stream>>>(Xb, Wvb, Vtb, 2);

  dim3 gs(16, 16, NBATCH);  // T/128, T/128, B
  scores_kernel<<<gs, blk, 0, stream>>>(Qb, Kb, S);

  softmax_kernel<<<NBATCH * T_SEQ, 256, 0, stream>>>(S, P);

  dim3 go(16, 8, NBATCH);  // T/128, D/128, B
  pv_kernel<<<go, blk, 0, stream>>>(P, Vtb, out);
}
